// SleepAudioFoundation_24893630447932
// MI455X (gfx1250) — matrix operand order repaired and checked by reading, not compiled, not run on hardware
//
#include <hip/hip_runtime.h>
#include <hip/hip_bf16.h>
#include <stdint.h>

// ---------------------------------------------------------------------------
// Audio transformer forward (AST-like) for MI455X / gfx1250, wave32 + WMMA.
// - all GEMMs via v_wmma_f32_16x16x32_bf16 (f32 accumulate)
// - A tiles (bf16 activations): GLOBAL_LOAD_ASYNC_TO_LDS_B128, double-buffered
// - B tiles (f32 weights): Tensor Data Mover (tensor_load_to_lds) into LDS,
//   converted f32->bf16 on the way into the transposed compute layout
// - attention: flash-style online softmax, DPP16 row_ror reductions,
//   V blocks async-staged into per-wave LDS
// All CDNA5-specific paths are __has_builtin-guarded with sync fallbacks.
// ---------------------------------------------------------------------------

typedef __attribute__((ext_vector_type(16))) __bf16 v16bf;
typedef __attribute__((ext_vector_type(8)))  __bf16 v8bf;
typedef __attribute__((ext_vector_type(8)))  float  v8f;
typedef __attribute__((ext_vector_type(4)))  unsigned int u32x4;
typedef __attribute__((ext_vector_type(8)))  int i32x8;
typedef __attribute__((ext_vector_type(4)))  int i32x4;
typedef int v4i_vs __attribute__((vector_size(16)));   // GCC-style int4 (async builtin pointee)

#define AS1 __attribute__((address_space(1)))
#define AS3 __attribute__((address_space(3)))

#if __has_builtin(__builtin_amdgcn_global_load_async_to_lds_b128)
#define HAVE_ASYNC 1
#else
#define HAVE_ASYNC 0
#endif
#if __has_builtin(__builtin_amdgcn_tensor_load_to_lds)
#define HAVE_TDM 1
#else
#define HAVE_TDM 0
#endif
#if __has_builtin(__builtin_amdgcn_update_dpp)
#define HAVE_DPP 1
#else
#define HAVE_DPP 0
#endif

constexpr int BATCH  = 4;
constexpr int N_MELS = 128;
constexpr int PSZ    = 16;
constexpr int NPATCH = 1024;
constexpr int DMODEL = 1024;
constexpr int NHEAD  = 16;
constexpr int NLAYER = 12;
constexpr int FFDIM  = 4096;
constexpr int HDIM   = 64;
constexpr int SEQ    = NPATCH + 1;     // 1025
constexpr int MR     = BATCH * SEQ;    // 4100 activation rows

// ---------------------------------------------------------------------------
// Async / TDM / DPP helpers
// ---------------------------------------------------------------------------
__device__ __forceinline__ void async_copy16(void* lds, const void* g) {
#if HAVE_ASYNC
  __builtin_amdgcn_global_load_async_to_lds_b128(
      (AS1 v4i_vs*)(void*)g, (AS3 v4i_vs*)lds, 0, 0);
#else
  *(v8bf*)lds = *(const v8bf*)g;
#endif
}

__device__ __forceinline__ void wait_async_all() {
#if HAVE_ASYNC
#if __has_builtin(__builtin_amdgcn_s_wait_asynccnt)
  __builtin_amdgcn_s_wait_asynccnt(0);
#else
  asm volatile("s_wait_asynccnt 0" ::: "memory");
#endif
#endif
}

__device__ __forceinline__ unsigned lds_byte_offset(const void* p) {
  return (unsigned)(uintptr_t)(AS3 const void*)p;
}

#if HAVE_TDM
// 2D tile of f32: rows x cols, row pitch `rstride` elements, -> LDS (packed).
// D# per CDNA5 ISA ch.8: group0 = {flags, lds_addr, global_addr, type=2},
// group1 = {data_size, tensor dims (== tile dims), tile dims, dim0 stride}.
__device__ __forceinline__ void tdm_load_tile_f32(unsigned lds_off,
                                                  const float* g, int rows,
                                                  int cols, int rstride) {
  unsigned long long ga = (unsigned long long)(uintptr_t)g;
  u32x4 g0;
  g0[0] = 1u;                                        // count=1, user D#
  g0[1] = lds_off;                                   // lds_addr (bytes)
  g0[2] = (unsigned)(ga & 0xFFFFFFFFull);            // global_addr[31:0]
  g0[3] = (unsigned)((ga >> 32) & 0x01FFFFFFull) | (2u << 30);  // [56:32]|type
  i32x8 g1;
  unsigned td0 = (unsigned)cols, td1 = (unsigned)rows;
  g1[0] = (int)(2u << 16);                           // data_size = 4B
  g1[1] = (int)((td0 & 0xFFFFu) << 16);              // tensor_dim0 lo16
  g1[2] = (int)(((td0 >> 16) & 0xFFFFu) | ((td1 & 0xFFFFu) << 16));
  g1[3] = (int)(((td1 >> 16) & 0xFFFFu) | ((td0 & 0xFFFFu) << 16)); // tile_dim0
  g1[4] = (int)(td1 & 0xFFFFu);                      // tile_dim1 (tile_dim2=0)
  g1[5] = (int)(unsigned)rstride;                    // tensor_dim0_stride lo32
  g1[6] = 0;
  g1[7] = 0;
  i32x4 gz = {0, 0, 0, 0};
#if __clang_major__ >= 23
  i32x8 gz8 = {0, 0, 0, 0, 0, 0, 0, 0};
  __builtin_amdgcn_tensor_load_to_lds(g0, g1, gz, gz, gz8, 0);
#else
  __builtin_amdgcn_tensor_load_to_lds(g0, g1, gz, gz, 0);
#endif
}
#endif

__device__ __forceinline__ void wait_tensor_all() {
#if HAVE_TDM
#if __has_builtin(__builtin_amdgcn_s_wait_tensorcnt)
  __builtin_amdgcn_s_wait_tensorcnt(0);
#else
  asm volatile("s_wait_tensorcnt 0" ::: "memory");
#endif
#endif
}

// reductions across a 16-lane group (DPP16 row_ror, rows==our groups in wave32)
__device__ __forceinline__ float red16_max(float v) {
#if HAVE_DPP
  v = fmaxf(v, __int_as_float(__builtin_amdgcn_update_dpp(0, __float_as_int(v), 0x128, 0xf, 0xf, true)));
  v = fmaxf(v, __int_as_float(__builtin_amdgcn_update_dpp(0, __float_as_int(v), 0x124, 0xf, 0xf, true)));
  v = fmaxf(v, __int_as_float(__builtin_amdgcn_update_dpp(0, __float_as_int(v), 0x122, 0xf, 0xf, true)));
  v = fmaxf(v, __int_as_float(__builtin_amdgcn_update_dpp(0, __float_as_int(v), 0x121, 0xf, 0xf, true)));
#else
  for (int m = 8; m >= 1; m >>= 1) v = fmaxf(v, __shfl_xor(v, m, 32));
#endif
  return v;
}
__device__ __forceinline__ float red16_sum(float v) {
#if HAVE_DPP
  v += __int_as_float(__builtin_amdgcn_update_dpp(0, __float_as_int(v), 0x128, 0xf, 0xf, true));
  v += __int_as_float(__builtin_amdgcn_update_dpp(0, __float_as_int(v), 0x124, 0xf, 0xf, true));
  v += __int_as_float(__builtin_amdgcn_update_dpp(0, __float_as_int(v), 0x122, 0xf, 0xf, true));
  v += __int_as_float(__builtin_amdgcn_update_dpp(0, __float_as_int(v), 0x121, 0xf, 0xf, true));
#else
  for (int m = 8; m >= 1; m >>= 1) v += __shfl_xor(v, m, 32);
#endif
  return v;
}

// ---------------------------------------------------------------------------
// GEMM: C[M,N] = A[M,K](bf16) @ Bw[K,N](f32) + bias.   128x64 block, 8 waves,
// wave = 32x32 (2x2 WMMA frags), BK=32, double-buffered LDS with async A
// staging and TDM B staging.
// EP=0: f32 out; EP=1: f32 out + residual; EP=2: bf16 QKV head scatter.
// ---------------------------------------------------------------------------
template <int EP>
__global__ __launch_bounds__(256) void gemm_kernel(
    const __bf16* __restrict__ A, const float* __restrict__ Bw,
    const float* __restrict__ bias, float* Cf, const float* Res,
    __bf16* Cb, int M, int N, int K, int Sdim) {
  __shared__ __align__(64) __bf16 As[2][128][32];
  __shared__ __align__(64) __bf16 Bs[2][64][32];     // transposed [n][k], bf16
#if HAVE_TDM
  __shared__ __align__(64) float Braw[2][32][64];    // raw f32 tile via TDM
#endif
  const int tid  = threadIdx.x;
  const int lane = tid & 31;
  const int wave = tid >> 5;
  const int wm   = wave >> 1;
  const int wn   = wave & 1;
  const int bm   = blockIdx.x * 128;
  const int bn   = blockIdx.y * 64;
  const int nl   = lane & 15;
  const int hi   = lane >> 4;

  v8f acc[2][2] = {};
  const int kTiles = K >> 5;

  auto stage_a = [&](int kt, int buf) {
    const int k0 = kt << 5;
#pragma unroll
    for (int c = 0; c < 2; ++c) {
      int idx = tid * 2 + c;
      int row = idx >> 2, col = (idx & 3) << 3;
      int gr = bm + row;
      gr = gr < M ? gr : M - 1;   // clamp: pad rows replicate, never stored
      async_copy16(&As[buf][row][col], A + (size_t)gr * K + k0 + col);
    }
  };
  auto stage_b = [&](int kt, int buf) {
#if HAVE_TDM
    if (wave == 0)
      tdm_load_tile_f32(lds_byte_offset(&Braw[buf][0][0]),
                        Bw + (size_t)(kt << 5) * N + bn, 32, 64, N);
#else
    (void)kt; (void)buf;
#endif
  };
  auto convert_b = [&](int kt, int buf) {
#pragma unroll
    for (int c = 0; c < 8; ++c) {
      int idx = tid + (c << 8);
      int kk = idx >> 6, nn = idx & 63;
#if HAVE_TDM
      float v = Braw[buf][kk][nn];
#else
      float v = Bw[(size_t)((kt << 5) + kk) * N + (bn + nn)];
#endif
      Bs[buf][nn][kk] = (__bf16)v;
    }
  };

  stage_a(0, 0);
  stage_b(0, 0);

  for (int kt = 0; kt < kTiles; ++kt) {
    const int cur = kt & 1;
    wait_async_all();                 // A tile kt landed (per-wave chunks)
    if (wave == 0) wait_tensor_all(); // B raw tile kt landed
    __syncthreads();
    if (kt + 1 < kTiles) {            // overlap next tile's DMA with compute
      stage_a(kt + 1, cur ^ 1);
      stage_b(kt + 1, cur ^ 1);
      __builtin_prefetch(Bw + (size_t)((kt + 1) << 5) * N + bn, 0, 1);
    }
    convert_b(kt, cur);               // f32 -> bf16, transpose into Bs[cur]
    __syncthreads();

    v16bf af[2], bf2[2];
#pragma unroll
    for (int i = 0; i < 2; ++i) {
      const __bf16* ar = &As[cur][wm * 32 + i * 16 + nl][0];
      v8bf lo = *(const v8bf*)(ar + hi * 16);
      v8bf hh = *(const v8bf*)(ar + hi * 16 + 8);
      af[i] = __builtin_shufflevector(lo, hh, 0, 1, 2, 3, 4, 5, 6, 7, 8, 9, 10,
                                      11, 12, 13, 14, 15);
    }
#pragma unroll
    for (int j = 0; j < 2; ++j)
      bf2[j] = *(const v16bf*)(&Bs[cur][wn * 32 + j * 16 + nl][hi * 16]);

#pragma unroll
    for (int i = 0; i < 2; ++i)
#pragma unroll
      for (int j = 0; j < 2; ++j)
        acc[i][j] = __builtin_amdgcn_wmma_f32_16x16x32_bf16(
            false, af[i], false, bf2[j], (short)0, acc[i][j], false, false);
  }

  // --- epilogue ---
#pragma unroll
  for (int i = 0; i < 2; ++i) {
#pragma unroll
    for (int j = 0; j < 2; ++j) {
      int   gn = bn + wn * 32 + j * 16 + nl;
      float bb = bias[gn];
#pragma unroll
      for (int e = 0; e < 8; ++e) {
        int gm = bm + wm * 32 + i * 16 + e + 8 * hi;
        if (gm < M) {
          float v = acc[i][j][e] + bb;
          if (EP == 0) {
            Cf[(size_t)gm * N + gn] = v;
          } else if (EP == 1) {
            Cf[(size_t)gm * N + gn] = Res[(size_t)gm * N + gn] + v;
          } else {
            int bq = gm / Sdim, sq = gm % Sdim;
            int head = gn >> 6, dd = gn & 63;
            Cb[(((size_t)(bq * NHEAD + head)) * Sdim + sq) * HDIM + dd] = (__bf16)v;
          }
        }
      }
    }
  }
}

// ---------------------------------------------------------------------------
// Flash attention: wave-per-16-row q tile, online softmax (DPP reductions),
// Q@K^T and P@V via WMMA.  V blocks async-staged into per-wave LDS; P goes
// D-frag -> LDS -> A-frag (intra-wave, DS in-order).
// ---------------------------------------------------------------------------
__global__ __launch_bounds__(128) void attn_kernel(
    const __bf16* __restrict__ Q, const __bf16* __restrict__ Km,
    const __bf16* __restrict__ Vm, __bf16* __restrict__ O) {
  __shared__ __align__(64) __bf16 Plds[4][16][32];
  __shared__ __align__(64) __bf16 Vlds[4][32][64];

  const int lane = threadIdx.x & 31;
  const int wave = threadIdx.x >> 5;
  const int nl   = lane & 15;
  const int hi   = lane >> 4;
  const int bh   = blockIdx.x;
  const int b    = bh >> 4;
  const int h    = bh & 15;
  const int qtile = blockIdx.y * 4 + wave;
  const int qbase = qtile * 16;
  const float scale = 0.125f;   // 1/sqrt(64)

  const __bf16* q  = Q  + (size_t)bh * SEQ * HDIM;
  const __bf16* kk = Km + (size_t)bh * SEQ * HDIM;
  const __bf16* vv = Vm + (size_t)bh * SEQ * HDIM;

  auto stage_v = [&](int kv) {
#pragma unroll
    for (int w = 0; w < 8; ++w) {
      int ci = lane + w * 32;            // 256 chunks of 16B = 32x64 bf16
      int row = ci >> 3, coloff = (ci & 7) << 3;
      int r = kv + row;
      r = r < SEQ ? r : SEQ - 1;         // clamp: P==0 for masked cols
      async_copy16(&Vlds[wave][row][coloff], vv + (size_t)r * HDIM + coloff);
    }
  };

  // Q fragments (DH=64 -> two K=32 steps)
  v16bf aq[2];
#pragma unroll
  for (int t = 0; t < 2; ++t) {
    v16bf a = {};
    if (qbase + nl < SEQ) {
      const __bf16* qp = q + (size_t)(qbase + nl) * HDIM + t * 32;
      v8bf lo = *(const v8bf*)(qp + hi * 16);
      v8bf hh = *(const v8bf*)(qp + hi * 16 + 8);
      a = __builtin_shufflevector(lo, hh, 0, 1, 2, 3, 4, 5, 6, 7, 8, 9, 10, 11,
                                  12, 13, 14, 15);
    }
    aq[t] = a;
  }

  v8f   oacc[4] = {};
  float mi[8], li[8];
#pragma unroll
  for (int e = 0; e < 8; ++e) { mi[e] = -1e30f; li[e] = 0.f; }

  for (int kv = 0; kv < SEQ; kv += 32) {
    stage_v(kv);   // async V block copy overlaps with score computation

    // ---- scores: two 16x16 frags covering kv..kv+31 ----
    v8f sfr[2];
#pragma unroll
    for (int f = 0; f < 2; ++f) {
      int kvrow = kv + f * 16 + nl;
      v8f s = {};
#pragma unroll
      for (int t = 0; t < 2; ++t) {
        v16bf bfrag = {};
        if (kvrow < SEQ)
          bfrag = *(const v16bf*)(kk + (size_t)kvrow * HDIM + t * 32 + hi * 16);
        s = __builtin_amdgcn_wmma_f32_16x16x32_bf16(
            false, aq[t], false, bfrag, (short)0, s, false, false);
      }
#pragma unroll
      for (int e = 0; e < 8; ++e)
        sfr[f][e] = (kvrow < SEQ) ? s[e] * scale : -1e30f;
    }

    // ---- online softmax (VALU DPP row reductions) ----
    float alpha[8];
#pragma unroll
    for (int e = 0; e < 8; ++e) {
      float vmax = red16_max(fmaxf(sfr[0][e], sfr[1][e]));
      float mnew = fmaxf(mi[e], vmax);
      alpha[e] = __expf(mi[e] - mnew);
      float p0 = __expf(sfr[0][e] - mnew);
      float p1 = __expf(sfr[1][e] - mnew);
      sfr[0][e] = p0; sfr[1][e] = p1;
      li[e] = li[e] * alpha[e] + red16_sum(p0 + p1);
      mi[e] = mnew;
    }
#pragma unroll
    for (int j = 0; j < 4; ++j)
#pragma unroll
      for (int e = 0; e < 8; ++e) oacc[j][e] *= alpha[e];

    // ---- P: D-frag -> LDS -> A-frag (intra-wave) ----
    __builtin_amdgcn_wave_barrier();
#pragma unroll
    for (int f = 0; f < 2; ++f)
#pragma unroll
      for (int e = 0; e < 8; ++e)
        Plds[wave][e + 8 * hi][f * 16 + nl] = (__bf16)sfr[f][e];
    asm volatile("s_wait_dscnt 0" ::: "memory");
    __builtin_amdgcn_wave_barrier();
    v16bf ap;
    {
      const __bf16* pr = &Plds[wave][nl][0];
      v8bf lo = *(const v8bf*)(pr + hi * 16);
      v8bf hh = *(const v8bf*)(pr + hi * 16 + 8);
      ap = __builtin_shufflevector(lo, hh, 0, 1, 2, 3, 4, 5, 6, 7, 8, 9, 10, 11,
                                   12, 13, 14, 15);
    }

    // ---- P @ V (V from per-wave LDS) ----
    wait_async_all();
#pragma unroll
    for (int j = 0; j < 4; ++j) {
      v16bf bv;
#pragma unroll
      for (int jj = 0; jj < 16; ++jj)
        bv[jj] = Vlds[wave][hi * 16 + jj][j * 16 + nl];
      oacc[j] = __builtin_amdgcn_wmma_f32_16x16x32_bf16(
          false, ap, false, bv, (short)0, oacc[j], false, false);
    }
    __builtin_amdgcn_wave_barrier();
  }

  // ---- epilogue: concat heads into [MR, DMODEL] bf16 ----
#pragma unroll
  for (int j = 0; j < 4; ++j) {
#pragma unroll
    for (int e = 0; e < 8; ++e) {
      int s = qbase + e + 8 * hi;
      if (s < SEQ) {
        float ov = oacc[j][e] / li[e];
        O[((size_t)b * SEQ + s) * DMODEL + h * HDIM + j * 16 + nl] = (__bf16)ov;
      }
    }
  }
}

// ---------------------------------------------------------------------------
// LayerNorm over D=1024, one block (256 thr) per row.
// MODE 0: bf16 out; MODE 1: f32 out; MODE 2: f32 out, patch-LN scatter.
// ---------------------------------------------------------------------------
template <int MODE>
__global__ __launch_bounds__(256) void ln_kernel(
    const float* __restrict__ X, const float* __restrict__ g,
    const float* __restrict__ be, void* __restrict__ Yv) {
  __shared__ float red[8];
  const int r = blockIdx.x;
  const int t = threadIdx.x;
  const float* x = X + (size_t)r * DMODEL;

  float v0[4];
  float s = 0.f;
#pragma unroll
  for (int i = 0; i < 4; ++i) { v0[i] = x[t + (i << 8)]; s += v0[i]; }
  s = red16_sum(s);
  s += __shfl_xor(s, 16, 32);
  if ((t & 31) == 0) red[t >> 5] = s;
  __syncthreads();
  float tot = 0.f;
#pragma unroll
  for (int i = 0; i < 8; ++i) tot += red[i];
  float mu = tot * (1.f / 1024.f);
  __syncthreads();

  float s2 = 0.f;
#pragma unroll
  for (int i = 0; i < 4; ++i) { float d = v0[i] - mu; s2 += d * d; }
  s2 = red16_sum(s2);
  s2 += __shfl_xor(s2, 16, 32);
  if ((t & 31) == 0) red[t >> 5] = s2;
  __syncthreads();
  float tv = 0.f;
#pragma unroll
  for (int i = 0; i < 8; ++i) tv += red[i];
  float rstd = rsqrtf(tv * (1.f / 1024.f) + 1e-5f);

#pragma unroll
  for (int i = 0; i < 4; ++i) {
    int   c = t + (i << 8);
    float o = (v0[i] - mu) * rstd * g[c] + be[c];
    if (MODE == 0) {
      ((__bf16*)Yv)[(size_t)r * DMODEL + c] = (__bf16)o;
    } else if (MODE == 1) {
      ((float*)Yv)[(size_t)r * DMODEL + c] = o;
    } else {
      int bb = r >> 10, p = r & 1023;
      ((float*)Yv)[((size_t)bb * SEQ + 1 + p) * DMODEL + c] = o;
    }
  }
}

// h[:,0,:] = cls_token
__global__ void cls_fill_kernel(const float* __restrict__ cls,
                                float* __restrict__ H) {
  int b = blockIdx.x, t = threadIdx.x;
#pragma unroll
  for (int i = 0; i < 4; ++i) {
    int c = t + (i << 8);
    H[(size_t)b * SEQ * DMODEL + c] = cls[c];
  }
}

// Ap[b*NP+p][m*16+j] = x[b][0][m][p*16+j]  (bf16)
__global__ void patch_extract_kernel(const float* __restrict__ x,
                                     __bf16* __restrict__ Ap) {
  size_t i = (size_t)blockIdx.x * 256 + threadIdx.x;
  if (i >= (size_t)BATCH * NPATCH * 2048) return;
  int    col = (int)(i & 2047);
  size_t rp  = i >> 11;
  int    p   = (int)(rp & 1023);
  int    b   = (int)(rp >> 10);
  int    m   = col >> 4, j = col & 15;
  Ap[i] = (__bf16)x[(((size_t)b * N_MELS + m) << 14) + p * PSZ + j];
}

// In-place RoPE on [B*H, S, 64] bf16; thread handles pair (d, d+32)
__global__ void rope_kernel(__bf16* __restrict__ q) {
  size_t i = (size_t)blockIdx.x * 256 + threadIdx.x;
  if (i >= (size_t)BATCH * NHEAD * SEQ * 32) return;
  int    d  = (int)(i & 31);
  size_t rs = i >> 5;
  int    s  = (int)(rs % SEQ);
  float inv = __powf(10000.f, -(float)d * (1.f / 32.f));
  float sn, cs;
  __sincosf((float)s * inv, &sn, &cs);
  __bf16* base = q + (rs << 6);
  float a = (float)base[d], bb = (float)base[d + 32];
  base[d]      = (__bf16)(a * cs - bb * sn);
  base[d + 32] = (__bf16)(bb * cs + a * sn);
}

// g = bf16( silu(t1) * t2 )
__global__ void swiglu_kernel(const float* __restrict__ t1,
                              const float* __restrict__ t2,
                              __bf16* __restrict__ g, size_t n) {
  size_t i = (size_t)blockIdx.x * 256 + threadIdx.x;
  if (i >= n) return;
  float a = t1[i];
  float si = a / (1.f + __expf(-a));
  g[i] = (__bf16)(si * t2[i]);
}

// ---------------------------------------------------------------------------
// Host-side orchestration
// ---------------------------------------------------------------------------
extern "C" void kernel_launch(void* const* d_in, const int* in_sizes, int n_in,
                              void* d_out, int out_size, void* d_ws,
                              size_t ws_size, hipStream_t stream) {
  (void)in_sizes; (void)n_in; (void)out_size; (void)ws_size;

  const float* x       = (const float*)d_in[0];
  const float* patch_w = (const float*)d_in[1];
  const float* patch_b = (const float*)d_in[2];
  const float* pln_g   = (const float*)d_in[3];
  const float* pln_b   = (const float*)d_in[4];
  const float* cls_t   = (const float*)d_in[5];
  const float* ln1_g   = (const float*)d_in[6];
  const float* ln1_b   = (const float*)d_in[7];
  const float* wq = (const float*)d_in[8];  const float* bq = (const float*)d_in[9];
  const float* wk = (const float*)d_in[10]; const float* bk = (const float*)d_in[11];
  const float* wv = (const float*)d_in[12]; const float* bv = (const float*)d_in[13];
  const float* wo = (const float*)d_in[14]; const float* bo = (const float*)d_in[15];
  const float* ln2_g = (const float*)d_in[16]; const float* ln2_b = (const float*)d_in[17];
  const float* w1 = (const float*)d_in[18]; const float* b1 = (const float*)d_in[19];
  const float* w2 = (const float*)d_in[20]; const float* b2 = (const float*)d_in[21];
  const float* w3 = (const float*)d_in[22]; const float* b3 = (const float*)d_in[23];
  const float* lnf_g = (const float*)d_in[24]; const float* lnf_b = (const float*)d_in[25];

  char*  ws  = (char*)d_ws;
  size_t off = 0;
  auto take = [&](size_t bytes) -> char* {
    char* p = ws + off;
    off += (bytes + 255) & ~(size_t)255;
    return p;
  };
  float*  hbuf = (float*)take((size_t)MR * DMODEL * 4);
  __bf16* ybuf = (__bf16*)take((size_t)MR * DMODEL * 2);
  __bf16* qbuf = (__bf16*)take((size_t)BATCH * NHEAD * SEQ * HDIM * 2);
  __bf16* kbuf = (__bf16*)take((size_t)BATCH * NHEAD * SEQ * HDIM * 2);
  __bf16* vbuf = (__bf16*)take((size_t)BATCH * NHEAD * SEQ * HDIM * 2);
  __bf16* obuf = (__bf16*)take((size_t)MR * DMODEL * 2);
  float*  t1   = (float*)take((size_t)MR * FFDIM * 4);
  float*  t2   = (float*)take((size_t)MR * FFDIM * 4);
  __bf16* gbuf = (__bf16*)take((size_t)MR * FFDIM * 2);
  __bf16* apatch = (__bf16*)gbuf;   // pre-layer only, aliases FFN scratch
  float*  p0     = t1;

  const int gM = (MR + 127) / 128;  // 33

  // ---- patch embedding ----
  {
    size_t tot = (size_t)BATCH * NPATCH * 2048;
    patch_extract_kernel<<<(int)((tot + 255) / 256), 256, 0, stream>>>(x, apatch);
  }
  gemm_kernel<0><<<dim3(4096 / 128, DMODEL / 64), 256, 0, stream>>>(
      apatch, patch_w, patch_b, p0, nullptr, nullptr, 4096, DMODEL, 2048, 0);
  ln_kernel<2><<<4096, 256, 0, stream>>>(p0, pln_g, pln_b, hbuf);
  cls_fill_kernel<<<BATCH, 256, 0, stream>>>(cls_t, hbuf);

  // ---- transformer layers ----
  for (int l = 0; l < NLAYER; ++l) {
    const float* Wq = wq + (size_t)l * DMODEL * DMODEL;
    const float* Wk = wk + (size_t)l * DMODEL * DMODEL;
    const float* Wv = wv + (size_t)l * DMODEL * DMODEL;
    const float* Wo = wo + (size_t)l * DMODEL * DMODEL;
    const float* W1 = w1 + (size_t)l * DMODEL * FFDIM;
    const float* W3 = w3 + (size_t)l * DMODEL * FFDIM;
    const float* W2 = w2 + (size_t)l * FFDIM * DMODEL;
    const float* Bq = bq + (size_t)l * DMODEL;
    const float* Bk = bk + (size_t)l * DMODEL;
    const float* Bv = bv + (size_t)l * DMODEL;
    const float* Bo = bo + (size_t)l * DMODEL;
    const float* B1 = b1 + (size_t)l * FFDIM;
    const float* B3 = b3 + (size_t)l * FFDIM;
    const float* B2 = b2 + (size_t)l * DMODEL;

    ln_kernel<0><<<MR, 256, 0, stream>>>(hbuf, ln1_g + (size_t)l * DMODEL,
                                         ln1_b + (size_t)l * DMODEL, ybuf);
    gemm_kernel<2><<<dim3(gM, DMODEL / 64), 256, 0, stream>>>(
        ybuf, Wq, Bq, nullptr, nullptr, qbuf, MR, DMODEL, DMODEL, SEQ);
    gemm_kernel<2><<<dim3(gM, DMODEL / 64), 256, 0, stream>>>(
        ybuf, Wk, Bk, nullptr, nullptr, kbuf, MR, DMODEL, DMODEL, SEQ);
    gemm_kernel<2><<<dim3(gM, DMODEL / 64), 256, 0, stream>>>(
        ybuf, Wv, Bv, nullptr, nullptr, vbuf, MR, DMODEL, DMODEL, SEQ);
    {
      size_t tot = (size_t)BATCH * NHEAD * SEQ * 32;
      int blocks = (int)((tot + 255) / 256);
      rope_kernel<<<blocks, 256, 0, stream>>>(qbuf);
      rope_kernel<<<blocks, 256, 0, stream>>>(kbuf);
    }
    attn_kernel<<<dim3(BATCH * NHEAD, 17), 128, 0, stream>>>(qbuf, kbuf, vbuf, obuf);
    gemm_kernel<1><<<dim3(gM, DMODEL / 64), 256, 0, stream>>>(
        obuf, Wo, Bo, hbuf, hbuf, nullptr, MR, DMODEL, DMODEL, 0);

    ln_kernel<0><<<MR, 256, 0, stream>>>(hbuf, ln2_g + (size_t)l * DMODEL,
                                         ln2_b + (size_t)l * DMODEL, ybuf);
    gemm_kernel<0><<<dim3(gM, FFDIM / 64), 256, 0, stream>>>(
        ybuf, W1, B1, t1, nullptr, nullptr, MR, FFDIM, DMODEL, 0);
    gemm_kernel<0><<<dim3(gM, FFDIM / 64), 256, 0, stream>>>(
        ybuf, W3, B3, t2, nullptr, nullptr, MR, FFDIM, DMODEL, 0);
    {
      size_t tot = (size_t)MR * FFDIM;
      swiglu_kernel<<<(int)((tot + 255) / 256), 256, 0, stream>>>(t1, t2, gbuf, tot);
    }
    gemm_kernel<1><<<dim3(gM, DMODEL / 64), 256, 0, stream>>>(
        gbuf, W2, B2, hbuf, hbuf, nullptr, MR, DMODEL, FFDIM, 0);
  }

  // ---- final layernorm -> d_out (f32 [B,S,D]) ----
  ln_kernel<1><<<MR, 256, 0, stream>>>(hbuf, lnf_g, lnf_b, (float*)d_out);
}
